// NegativeMaskToPoints_73151882985670
// MI455X (gfx1250) — matrix operand order repaired and checked by reading, not compiled, not run on hardware
//
#include <hip/hip_runtime.h>
#include <hip/hip_bf16.h>

// Problem constants (from reference)
#define NMASK 64
#define IMG_H 1024
#define IMG_W 1024
#define HW (IMG_H * IMG_W)
#define NPTS_TOTAL 5

typedef __attribute__((ext_vector_type(16))) _Float16 v16h;
typedef __attribute__((ext_vector_type(2)))  __fp16   hp2;   // cvt_pkrtz result type
typedef __attribute__((ext_vector_type(8)))  float    v8f;
typedef __attribute__((ext_vector_type(4)))  float    v4f;

union AFrag { v16h v; hp2 h[8]; };

// ---------------------------------------------------------------------------
// Workspace layout (bytes):
//   [0,    1024)  per-mask accumulators: {count, sum_y, sum_x, pad} x 64 (f32)
//   [1024, 3072)  argmax slots: 4 iterations x 64 masks (u64)
//   [3072, 5632)  chosen points: 64 masks x 5 points x {y, x} (f32)
// ---------------------------------------------------------------------------

__global__ void nm2p_init(float* __restrict__ acc,
                          unsigned long long* __restrict__ slots,
                          float* __restrict__ out) {
    int tid = threadIdx.x;
    if (tid < NMASK * 4) acc[tid] = 0.0f;
    if (tid < 4 * NMASK) slots[tid] = 0ull;
    // labels: 320 int32 zeros appended after 640 point floats
    if (tid < NMASK * NPTS_TOTAL)
        ((unsigned int*)out)[NMASK * NPTS_TOTAL * 2 + tid] = 0u;
}

// ---------------------------------------------------------------------------
// Pass 1: centroid reduction via v_wmma_f32_16x16x32_f16.
// One wave handles a 16-row x 1024-col strip of one mask.
// A[M][K] = mask value (0/1, exact in f16); B col0 = 1 (count),
// B col1 = global x coordinate of K (exact in f16, +32/step, double-buffered
// to avoid WMMA->VALU WAR hazard NOPs).
// C (f32) accumulates: col0 = per-row count, col1 = per-row sum(m*x).
// Software-pipelined: next tile's loads issue before current tile's WMMA.
// ---------------------------------------------------------------------------
__device__ __forceinline__ void nm2p_cvt_tile(AFrag& af, v4f f0, v4f f1, v4f f2, v4f f3) {
    af.h[0] = __builtin_amdgcn_cvt_pkrtz(f0[0], f0[1]);
    af.h[1] = __builtin_amdgcn_cvt_pkrtz(f0[2], f0[3]);
    af.h[2] = __builtin_amdgcn_cvt_pkrtz(f1[0], f1[1]);
    af.h[3] = __builtin_amdgcn_cvt_pkrtz(f1[2], f1[3]);
    af.h[4] = __builtin_amdgcn_cvt_pkrtz(f2[0], f2[1]);
    af.h[5] = __builtin_amdgcn_cvt_pkrtz(f2[2], f2[3]);
    af.h[6] = __builtin_amdgcn_cvt_pkrtz(f3[0], f3[1]);
    af.h[7] = __builtin_amdgcn_cvt_pkrtz(f3[2], f3[3]);
}

__global__ void nm2p_centroid_wmma(const float* __restrict__ masks,
                                   float* __restrict__ acc) {
    const int lane = threadIdx.x & 31;
    const int wid  = threadIdx.x >> 5;
    const int strip_g = blockIdx.x * 8 + wid;        // 4096 strips total
    const int m        = strip_g >> 6;               // mask id (64 strips/mask)
    const int strip    = strip_g & 63;
    const int row_base = strip * 16;

    const int r  = lane & 15;                        // A row M within tile
    const int hf = lane >> 4;                        // lane-half (K interleave)
    const int c0 = 8 * hf;                           // first K run offset

    const float* __restrict__ rowp =
        masks + ((size_t)m * IMG_H + (row_base + r)) * IMG_W + c0;

    // B fragment: element e of lane holds K = kh + e (+8 for e >= 8), column n = lane & 15
    const int n  = lane & 15;
    const int kh = hf * 8;
    v16h bcur, binc;
    #pragma unroll
    for (int e = 0; e < 16; ++e) {
        float bv = (n == 0) ? 1.0f : ((n == 1) ? (float)(kh + e + (e >= 8 ? 8 : 0)) : 0.0f);
        bcur[e] = (_Float16)bv;
        binc[e] = (_Float16)((n == 1) ? 32.0f : 0.0f);
    }

    v8f cacc = {};

    // Prologue: load tile for xb = 0
    v4f f0 = *(const v4f*)(rowp + 0);
    v4f f1 = *(const v4f*)(rowp + 4);
    v4f f2 = *(const v4f*)(rowp + 16);
    v4f f3 = *(const v4f*)(rowp + 20);

    for (int xb = 0; xb < IMG_W - 32; xb += 32) {
        AFrag af;
        nm2p_cvt_tile(af, f0, f1, f2, f3);
        // Issue next tile's loads before consuming this tile in WMMA
        const float* np = rowp + xb + 32;
        f0 = *(const v4f*)(np + 0);
        f1 = *(const v4f*)(np + 4);
        f2 = *(const v4f*)(np + 16);
        f3 = *(const v4f*)(np + 20);
        v16h bnext = bcur + binc;   // writes fresh regs: no WAR on WMMA sources
        cacc = __builtin_amdgcn_wmma_f32_16x16x32_f16(
            false, af.v, false, bcur, (short)0, cacc, false, false);
        bcur = bnext;
    }
    {   // Epilogue: last tile
        AFrag af;
        nm2p_cvt_tile(af, f0, f1, f2, f3);
        cacc = __builtin_amdgcn_wmma_f32_16x16x32_f16(
            false, af.v, false, bcur, (short)0, cacc, false, false);
    }

    // Extract: C VGPR j -> lanes 0-15 row j, lanes 16-31 row j+8
    float cnt = 0.0f, sx = 0.0f, sy = 0.0f;
    #pragma unroll
    for (int j = 0; j < 8; ++j) {
        float cA = __shfl(cacc[j], 0, 32);
        float cB = __shfl(cacc[j], 16, 32);
        float xA = __shfl(cacc[j], 1, 32);
        float xB = __shfl(cacc[j], 17, 32);
        cnt += cA + cB;
        sx  += xA + xB;
        sy  += (float)(row_base + j) * cA + (float)(row_base + 8 + j) * cB;
    }
    if (lane == 0) {
        __hip_atomic_fetch_add(&acc[m * 4 + 0], cnt, __ATOMIC_RELAXED, __HIP_MEMORY_SCOPE_AGENT);
        __hip_atomic_fetch_add(&acc[m * 4 + 1], sy,  __ATOMIC_RELAXED, __HIP_MEMORY_SCOPE_AGENT);
        __hip_atomic_fetch_add(&acc[m * 4 + 2], sx,  __ATOMIC_RELAXED, __HIP_MEMORY_SCOPE_AGENT);
    }
}

__global__ void nm2p_centroid_final(const float* __restrict__ acc,
                                    float* __restrict__ pts,
                                    float* __restrict__ out) {
    int m = threadIdx.x;
    if (m < NMASK) {
        float cnt = acc[m * 4 + 0];
        float cy  = acc[m * 4 + 1] / cnt;
        float cx  = acc[m * 4 + 2] / cnt;
        pts[m * 10 + 0] = cy;
        pts[m * 10 + 1] = cx;
        out[m * 10 + 0] = cx;   // (m*5 + 0)*2, output order (x, y)
        out[m * 10 + 1] = cy;
    }
}

// ---------------------------------------------------------------------------
// FPS iteration: argmax over min-distance to chosen points.
// Packed u64 = (float_bits(d) << 32) | ~idx  -> max picks largest d,
// then lowest linear index (matches jnp.argmax first-tie semantics).
// Vectorized: each thread processes 4 consecutive pixels per b128 load.
// ---------------------------------------------------------------------------
__global__ void nm2p_fps_reduce(const float* __restrict__ masks,
                                const float* __restrict__ pts,
                                unsigned long long* __restrict__ slot,
                                int npts) {
    const int m = blockIdx.y;
    const float* __restrict__ mp = masks + ((size_t)m << 20);

    float py[NPTS_TOTAL], px[NPTS_TOTAL];
    for (int j = 0; j < npts; ++j) {
        py[j] = pts[m * 10 + 2 * j + 0];
        px[j] = pts[m * 10 + 2 * j + 1];
    }

    unsigned long long best = 0ull;
    // 128 blocks x 256 threads x 4 px = 131072 px per sweep; 8 sweeps = 1M px.
    int p = (blockIdx.x * 256 + threadIdx.x) * 4;
    #pragma unroll 2
    for (int i = 0; i < 8; ++i, p += 128 * 256 * 4) {
        v4f mv = *(const v4f*)(mp + p);
        float y  = (float)(p >> 10);          // quad stays within one row
        float x0 = (float)(p & 1023);
        #pragma unroll
        for (int c = 0; c < 4; ++c) {
            if (mv[c] > 0.0f) {
                float x = x0 + (float)c;
                float d2 = 3.4e38f;
                for (int j = 0; j < npts; ++j) {
                    float dy = y - py[j], dx = x - px[j];
                    d2 = fminf(d2, dy * dy + dx * dx);
                }
                float d = sqrtf(d2);  // monotone: min of sqrt == sqrt of min
                unsigned long long pk =
                    ((unsigned long long)__float_as_uint(d) << 32) |
                    (unsigned int)(~(unsigned int)(p + c));
                if (pk > best) best = pk;
            }
        }
    }

    #pragma unroll
    for (int off = 16; off > 0; off >>= 1) {
        unsigned long long o = __shfl_xor(best, off, 32);
        if (o > best) best = o;
    }
    __shared__ unsigned long long sm[8];
    int lane = threadIdx.x & 31, wid = threadIdx.x >> 5;
    if (lane == 0) sm[wid] = best;
    __syncthreads();
    if (wid == 0) {
        best = (lane < 8) ? sm[lane] : 0ull;
        #pragma unroll
        for (int off = 4; off > 0; off >>= 1) {
            unsigned long long o = __shfl_xor(best, off, 32);
            if (o > best) best = o;
        }
        if (lane == 0) atomicMax(&slot[m], best);
    }
}

__global__ void nm2p_fps_select(const unsigned long long* __restrict__ slot,
                                float* __restrict__ pts,
                                float* __restrict__ out,
                                int it) {
    int m = threadIdx.x;
    if (m < NMASK) {
        unsigned int idx = ~(unsigned int)(slot[m] & 0xffffffffull);
        float y = (float)(idx >> 10);
        float x = (float)(idx & 1023);
        pts[m * 10 + 2 * it + 0] = y;
        pts[m * 10 + 2 * it + 1] = x;
        out[(m * NPTS_TOTAL + it) * 2 + 0] = x;
        out[(m * NPTS_TOTAL + it) * 2 + 1] = y;
    }
}

// ---------------------------------------------------------------------------
extern "C" void kernel_launch(void* const* d_in, const int* in_sizes, int n_in,
                              void* d_out, int out_size, void* d_ws, size_t ws_size,
                              hipStream_t stream) {
    const float* masks = (const float*)d_in[0];
    float* out = (float*)d_out;

    float* acc = (float*)d_ws;                                             // 1 KB
    unsigned long long* slots = (unsigned long long*)((char*)d_ws + 1024); // 2 KB
    float* pts = (float*)((char*)d_ws + 3072);                             // 2.5 KB

    nm2p_init<<<1, 512, 0, stream>>>(acc, slots, out);

    // 64 masks x 64 strips = 4096 waves; 8 waves (256 threads) per block.
    nm2p_centroid_wmma<<<512, 256, 0, stream>>>(masks, acc);
    nm2p_centroid_final<<<1, 64, 0, stream>>>(acc, pts, out);

    dim3 fgrid(128, NMASK);
    for (int it = 1; it < NPTS_TOTAL; ++it) {
        nm2p_fps_reduce<<<fgrid, 256, 0, stream>>>(masks, pts,
                                                   slots + (size_t)(it - 1) * NMASK, it);
        nm2p_fps_select<<<1, 64, 0, stream>>>(slots + (size_t)(it - 1) * NMASK,
                                              pts, out, it);
    }
}
